// YOLOTOS_70480413328023
// MI455X (gfx1250) — matrix operand order repaired and checked by reading, not compiled, not run on hardware
//
#include <hip/hip_runtime.h>
#include <hip/hip_bf16.h>

// ---------------------------------------------------------------------------
// CDNA5 (gfx1250) WMMA pipeline for VGG16 + GRU enc/dec detection head.
// - All matmuls via V_WMMA_F32_16X16X32_F16 (f16 A/B, f32 accumulate)
// - f16 activations + pre-padded f16 operands: GEMM staging is guard-free
//   aligned b128 copies via global_load_async_to_lds_b128 (ASYNCcnt)
// - Double-buffered LDS tiles: next tile's async DMA (and im2col gather)
//   overlaps the current tile's WMMAs; waits use s_wait_asynccnt(N) to keep
//   the next tile's copies in flight.
// ---------------------------------------------------------------------------

typedef __attribute__((ext_vector_type(16))) _Float16 v16h;
typedef __attribute__((ext_vector_type(8)))  _Float16 v8h;
typedef __attribute__((ext_vector_type(8)))  float    v8f;

#if defined(__gfx1250__) && __has_builtin(__builtin_amdgcn_global_load_async_to_lds_b128) && __has_builtin(__builtin_amdgcn_s_wait_asynccnt)
#define ASYNC_LDS 1
typedef int v4i_ __attribute__((vector_size(16)));
typedef __attribute__((address_space(1))) v4i_* gv4i_p;   // global int4*
typedef __attribute__((address_space(3))) v4i_* lv4i_p;   // LDS int4*
#define ASYNC_COPY_B128(gsrc, ldst) \
  __builtin_amdgcn_global_load_async_to_lds_b128((gv4i_p)(gsrc), (lv4i_p)(ldst), 0, 0)
#define ASYNC_WAIT_N(more, depth) \
  do { if (more) __builtin_amdgcn_s_wait_asynccnt(depth); \
       else      __builtin_amdgcn_s_wait_asynccnt(0); } while (0)
#else
#define ASYNC_LDS 0
#define ASYNC_COPY_B128(gsrc, ldst) (*(v8h*)(ldst) = *(const v8h*)(gsrc))
#define ASYNC_WAIT_N(more, depth) ((void)0)
#endif

__device__ __forceinline__ v16h mk16(v8h lo, v8h hi) {
  v16h r;
#pragma unroll
  for (int i = 0; i < 8; ++i) { r[i] = lo[i]; r[8 + i] = hi[i]; }
  return r;
}

// Branchless guarded load (clamped index + select) for the im2col gather.
__device__ __forceinline__ _Float16 ldh(const _Float16* __restrict__ p, size_t i, bool ok) {
  _Float16 v = p[ok ? i : 0];
  return ok ? v : (_Float16)0.f;
}

__device__ __forceinline__ float sigmoidf_(float x) { return 1.f / (1.f + expf(-x)); }

// ---------------------------------------------------------------------------
// GEMM on pre-padded f16 operands:
//   C[M,N] = act(A16[Mpad][K32] @ W16[Npad][K32]^T + bias)
// A16 rows >= ceil32(M) (pad rows zero), lda16 multiple of 32.
// W16 rows >= ceil64(N) (pad rows zero), row stride K32 (multiple of 32).
// Optional f32 output C and/or f16 output C16 (for chained GEMMs).
// Block 128 thr = 4 waves; tile 32(M) x 64(N); 2 WMMAs/wave/K-step sharing B.
// Double-buffered: tile k+1's 3 async copies/thread overlap tile k's WMMAs.
// ---------------------------------------------------------------------------
__global__ __launch_bounds__(128)
void gemm16_wmma(const _Float16* __restrict__ A16, int lda16,
                 const _Float16* __restrict__ W16, const float* __restrict__ bias,
                 float* __restrict__ C, int ldc,
                 _Float16* __restrict__ C16, int ldc16,
                 int M, int N, int K32, int act) {
  __shared__ __align__(16) _Float16 As[2][32][40];   // 80B rows (bank-friendly)
  __shared__ __align__(16) _Float16 Ws[2][64][40];
  const int tid  = threadIdx.x;
  const int wave = tid >> 5;
  const int lane = tid & 31;
  const int n    = lane & 15;
  const int khf  = lane >> 4;
  const int m0    = blockIdx.x * 32;
  const int nBase = blockIdx.y * 64;
  const int nk    = K32 >> 5;

  auto issueTile = [&](int kt, int buf) {
    int k0 = kt * 32;
    {
      int row = tid >> 2, q = (tid & 3) * 8;
      ASYNC_COPY_B128(A16 + (size_t)(m0 + row) * lda16 + k0 + q, &As[buf][row][q]);
    }
#pragma unroll
    for (int i = 0; i < 2; ++i) {
      int cidx = tid * 2 + i;
      int row = cidx >> 2, q = (cidx & 3) * 8;
      ASYNC_COPY_B128(W16 + (size_t)(nBase + row) * K32 + k0 + q, &Ws[buf][row][q]);
    }
  };

  v8f acc0 = {}, acc1 = {};
  issueTile(0, 0);
  for (int kt = 0; kt < nk; ++kt) {
    const int  cur  = kt & 1;
    const bool more = (kt + 1 < nk);
    if (more) issueTile(kt + 1, cur ^ 1);      // DMA for next tile stays in flight
    ASYNC_WAIT_N(more, 3);                     // only current tile's 3 copies must land
    __syncthreads();

    int ar0 = lane & 15, ar1 = 16 + ar0;
    v16h af0 = mk16(*(const v8h*)&As[cur][ar0][khf * 8], *(const v8h*)&As[cur][ar0][16 + khf * 8]);
    v16h af1 = mk16(*(const v8h*)&As[cur][ar1][khf * 8], *(const v8h*)&As[cur][ar1][16 + khf * 8]);
    int wrow = wave * 16 + n;
    v16h bf = mk16(*(const v8h*)&Ws[cur][wrow][khf * 8], *(const v8h*)&Ws[cur][wrow][16 + khf * 8]);

    acc0 = __builtin_amdgcn_wmma_f32_16x16x32_f16(false, af0, false, bf, (short)0, acc0, false, false);
    acc1 = __builtin_amdgcn_wmma_f32_16x16x32_f16(false, af1, false, bf, (short)0, acc1, false, false);
    __syncthreads();
  }

  const int col = nBase + wave * 16 + n;
#pragma unroll
  for (int r = 0; r < 8; ++r) {
    int gm0 = m0 + r + 8 * khf;
    int gm1 = gm0 + 16;
    float bv = (bias && col < N) ? bias[col] : 0.f;
    float v0 = acc0[r] + bv, v1 = acc1[r] + bv;
    if (act == 1)      { v0 = fmaxf(v0, 0.f);  v1 = fmaxf(v1, 0.f); }
    else if (act == 2) { v0 = sigmoidf_(v0);   v1 = sigmoidf_(v1); }
    if (col < N) {
      if (C) {
        if (gm0 < M) C[(size_t)gm0 * ldc + col] = v0;
        if (gm1 < M) C[(size_t)gm1 * ldc + col] = v1;
      }
      if (C16) {
        if (gm0 < M) C16[(size_t)gm0 * ldc16 + col] = (_Float16)v0;
        if (gm1 < M) C16[(size_t)gm1 * ldc16 + col] = (_Float16)v1;
      }
    }
  }
}

// ---------------------------------------------------------------------------
// 3x3 conv (pad=1), implicit GEMM, f16 in / f16 out, fused bias+ReLU.
// Block tile: 64 Cout x 32 positions; 2 WMMAs/wave/K-step sharing the weight
// fragment.  Double-buffered: next weight tile via async DMA and next im2col
// gather held in registers while current tile's WMMAs run.
// ---------------------------------------------------------------------------
__global__ __launch_bounds__(128)
void conv3x3_wmma(const _Float16* __restrict__ in, const _Float16* __restrict__ w16,
                  const float* __restrict__ bias, _Float16* __restrict__ out,
                  int Cin, int Cout, int H, int W, int K, int K32) {
  __shared__ __align__(16) _Float16 Bs[2][32][40];
  __shared__ __align__(16) _Float16 As16[2][64][40];
  const int HW = H * W;
  const int tid  = threadIdx.x;
  const int wave = tid >> 5;
  const int lane = tid & 31;
  const int n    = lane & 15;
  const int khf  = lane >> 4;
  const int posBase  = blockIdx.x * 32;
  const int coutBase = blockIdx.y * 64;
  const int b = blockIdx.z;
  const size_t inB = (size_t)b * Cin * HW;
  const int nk = (K + 31) >> 5;

  const int p = tid >> 2, c = (tid & 3) * 8;
  const int pos = posBase + p;
  const int y = pos / W, x = pos - y * W;

  auto gatherTile = [&](int kt) -> v8h {   // branchless im2col halo gather
    int k0 = kt * 32;
    v8h vals;
#pragma unroll
    for (int i = 0; i < 8; ++i) {
      int gk = k0 + c + i;
      int cin = gk / 9, r = gk - cin * 9;
      int kh = r / 3, kw = r - kh * 3;
      int iy = y + kh - 1, ix = x + kw - 1;
      bool ok = (gk < K) && (pos < HW) && iy >= 0 && iy < H && ix >= 0 && ix < W;
      vals[i] = ldh(in, inB + (size_t)cin * HW + (size_t)iy * W + ix, ok);
    }
    return vals;
  };
  auto issueW = [&](int kt, int buf) {
    int k0 = kt * 32;
#pragma unroll
    for (int i = 0; i < 2; ++i) {
      int cidx = tid * 2 + i;
      int row = cidx >> 2, q = (cidx & 3) * 8;
      ASYNC_COPY_B128(w16 + (size_t)(coutBase + row) * K32 + k0 + q, &As16[buf][row][q]);
    }
  };

  v8f acc0 = {}, acc1 = {};
  issueW(0, 0);
  { v8h v0 = gatherTile(0); *(v8h*)&Bs[0][p][c] = v0; }
  for (int kt = 0; kt < nk; ++kt) {
    const int  cur  = kt & 1;
    const bool more = (kt + 1 < nk);
    v8h pre = {};
    if (more) { issueW(kt + 1, cur ^ 1); pre = gatherTile(kt + 1); }
    ASYNC_WAIT_N(more, 2);
    __syncthreads();

    int arow = wave * 16 + (lane & 15);
    v16h af  = mk16(*(const v8h*)&As16[cur][arow][khf * 8], *(const v8h*)&As16[cur][arow][16 + khf * 8]);
    v16h bf0 = mk16(*(const v8h*)&Bs[cur][n][khf * 8],      *(const v8h*)&Bs[cur][n][16 + khf * 8]);
    v16h bf1 = mk16(*(const v8h*)&Bs[cur][16 + n][khf * 8], *(const v8h*)&Bs[cur][16 + n][16 + khf * 8]);

    acc0 = __builtin_amdgcn_wmma_f32_16x16x32_f16(false, af, false, bf0, (short)0, acc0, false, false);
    acc1 = __builtin_amdgcn_wmma_f32_16x16x32_f16(false, af, false, bf1, (short)0, acc1, false, false);

    if (more) *(v8h*)&Bs[cur ^ 1][p][c] = pre;   // buffer dead since kt-1's end barrier
    __syncthreads();
  }

  const int coutW = coutBase + wave * 16;
  int p0 = posBase + n, p1 = posBase + 16 + n;
#pragma unroll
  for (int r = 0; r < 8; ++r) {
    int cout = coutW + r + 8 * khf;
    size_t obase = ((size_t)b * Cout + cout) * HW;
    if (p0 < HW) out[obase + p0] = (_Float16)fmaxf(acc0[r] + bias[cout], 0.f);
    if (p1 < HW) out[obase + p1] = (_Float16)fmaxf(acc1[r] + bias[cout], 0.f);
  }
}

// ---------------------------------------------------------------------------
// 1x1 conv (projection), f16 in, writes f16 *directly into combined16*
// (B,64 tokens,3264-stride rows, valid width 3234):
//   clsPack==0: row = b*64 + cout,        off = pos            (det tokens)
//   clsPack==1: row = b*64 + (cout>>1),   off = 3136 + 49*(cout&1) + pos
// ---------------------------------------------------------------------------
__global__ __launch_bounds__(128)
void conv1x1_wmma(const _Float16* __restrict__ in, const _Float16* __restrict__ w16,
                  const float* __restrict__ bias, _Float16* __restrict__ out,
                  int Cin, int Cout, int HW, int K32, int clsPack) {
  __shared__ __align__(16) _Float16 Bs[2][32][40];
  __shared__ __align__(16) _Float16 As16[2][64][40];
  const int tid  = threadIdx.x;
  const int wave = tid >> 5;
  const int lane = tid & 31;
  const int n    = lane & 15;
  const int khf  = lane >> 4;
  const int posBase  = blockIdx.x * 32;
  const int coutBase = blockIdx.y * 64;
  const int b = blockIdx.z;
  const int nk = Cin >> 5;

  const int p = tid >> 2, c = (tid & 3) * 8;
  const int pos = posBase + p;

  auto gatherTile = [&](int kt) -> v8h {
    int k0 = kt * 32;
    v8h vals;
#pragma unroll
    for (int i = 0; i < 8; ++i) {
      int gk = k0 + c + i;
      bool ok = (gk < Cin) && (pos < HW);
      vals[i] = ldh(in, ((size_t)b * Cin + gk) * HW + pos, ok);
    }
    return vals;
  };
  auto issueW = [&](int kt, int buf) {
    int k0 = kt * 32;
#pragma unroll
    for (int i = 0; i < 2; ++i) {
      int cidx = tid * 2 + i;
      int row = cidx >> 2, q = (cidx & 3) * 8;
      ASYNC_COPY_B128(w16 + (size_t)(coutBase + row) * K32 + k0 + q, &As16[buf][row][q]);
    }
  };

  v8f acc0 = {}, acc1 = {};
  issueW(0, 0);
  { v8h v0 = gatherTile(0); *(v8h*)&Bs[0][p][c] = v0; }
  for (int kt = 0; kt < nk; ++kt) {
    const int  cur  = kt & 1;
    const bool more = (kt + 1 < nk);
    v8h pre = {};
    if (more) { issueW(kt + 1, cur ^ 1); pre = gatherTile(kt + 1); }
    ASYNC_WAIT_N(more, 2);
    __syncthreads();

    int arow = wave * 16 + (lane & 15);
    v16h af  = mk16(*(const v8h*)&As16[cur][arow][khf * 8], *(const v8h*)&As16[cur][arow][16 + khf * 8]);
    v16h bf0 = mk16(*(const v8h*)&Bs[cur][n][khf * 8],      *(const v8h*)&Bs[cur][n][16 + khf * 8]);
    v16h bf1 = mk16(*(const v8h*)&Bs[cur][16 + n][khf * 8], *(const v8h*)&Bs[cur][16 + n][16 + khf * 8]);

    acc0 = __builtin_amdgcn_wmma_f32_16x16x32_f16(false, af, false, bf0, (short)0, acc0, false, false);
    acc1 = __builtin_amdgcn_wmma_f32_16x16x32_f16(false, af, false, bf1, (short)0, acc1, false, false);

    if (more) *(v8h*)&Bs[cur ^ 1][p][c] = pre;
    __syncthreads();
  }

  const int coutW = coutBase + wave * 16;
  int p0 = posBase + n, p1 = posBase + 16 + n;
#pragma unroll
  for (int r = 0; r < 8; ++r) {
    int cout = coutW + r + 8 * khf;
    float v0 = acc0[r] + bias[cout];
    float v1 = acc1[r] + bias[cout];
    size_t row, offs;
    if (clsPack) { row = (size_t)b * 64 + (cout >> 1); offs = 3136 + 49 * (cout & 1); }
    else         { row = (size_t)b * 64 + cout;        offs = 0; }
    size_t tok = row * 3264 + offs;
    if (p0 < HW) out[tok + p0] = (_Float16)v0;
    if (p1 < HW) out[tok + p1] = (_Float16)v1;
  }
}

// 2x2 stride-2 pooling (max or avg), NCHW, f16.
__global__ void pool2_kernel(const _Float16* __restrict__ in, _Float16* __restrict__ out,
                             int C, int Hin, int Win, int isMax) {
  int Ho = Hin >> 1, Wo = Win >> 1;
  int total = 4 * C * Ho * Wo;
  int idx = blockIdx.x * blockDim.x + threadIdx.x;
  if (idx >= total) return;
  int wo = idx % Wo; int t = idx / Wo;
  int ho = t % Ho;   t /= Ho;
  int c  = t % C;    int b = t / C;
  size_t base = (((size_t)b * C + c) * Hin + 2 * ho) * Win + 2 * wo;
  float a0 = (float)in[base], a1 = (float)in[base + 1];
  float a2 = (float)in[base + Win], a3 = (float)in[base + Win + 1];
  float v = isMax ? fmaxf(fmaxf(a0, a1), fmaxf(a2, a3)) : 0.25f * (a0 + a1 + a2 + a3);
  out[idx] = (_Float16)v;
}

// GRU elementwise combine; writes f32 h and an f16 mirror (padded A buffer).
__global__ void gru_kernel(const float* __restrict__ xp, long xps,
                           const float* __restrict__ hp,
                           const float* __restrict__ hprev,
                           float* __restrict__ hnew, _Float16* __restrict__ hnew16,
                           float* __restrict__ out2, long o2s) {
  int idx = blockIdx.x * blockDim.x + threadIdx.x;
  if (idx >= 4 * 512) return;
  int b = idx >> 9, j = idx & 511;
  const float* xr = xp + (size_t)b * xps;
  const float* hr = hp + (size_t)b * 1536;
  float Xr = xr[j], Xz = xr[512 + j], Xn = xr[1024 + j];
  float Hr = hr[j], Hz = hr[512 + j], Hn = hr[1024 + j];
  float hv = hprev ? hprev[idx] : 0.f;
  float r = sigmoidf_(Xr + Hr);
  float z = sigmoidf_(Xz + Hz);
  float nn = tanhf(Xn + r * Hn);
  float h = (1.f - z) * nn + z * hv;
  hnew[idx] = h;
  if (hnew16) hnew16[idx] = (_Float16)h;
  if (out2) out2[(size_t)b * o2s + j] = h;
}

// Luong attention over 64 encoder outputs; writes f32 + f16 mirrors.
__global__ __launch_bounds__(256)
void attn_kernel(const float* __restrict__ q, const float* __restrict__ enc,
                 float* __restrict__ hout, _Float16* __restrict__ hout16,
                 float* __restrict__ hid, _Float16* __restrict__ hid16, long hidStride) {
  __shared__ float sc[64];
  __shared__ float mx, sm;
  int b = blockIdx.x, t = threadIdx.x;
  if (t < 64) {
    const float* qa = q + (size_t)b * 512;
    const float* e  = enc + ((size_t)b * 64 + t) * 512;
    float d = 0.f;
    for (int k = 0; k < 512; ++k) d += qa[k] * e[k];
    sc[t] = d;
  }
  __syncthreads();
  if (t == 0) { float m = sc[0]; for (int s = 1; s < 64; ++s) m = fmaxf(m, sc[s]); mx = m; }
  __syncthreads();
  if (t < 64) sc[t] = expf(sc[t] - mx);
  __syncthreads();
  if (t == 0) { float s = 0.f; for (int i = 0; i < 64; ++i) s += sc[i]; sm = s; }
  __syncthreads();
  float inv = 1.f / sm;
  for (int h = t; h < 512; h += blockDim.x) {
    float a = 0.f;
    for (int s = 0; s < 64; ++s) a += sc[s] * enc[((size_t)b * 64 + s) * 512 + h];
    a *= inv;
    hout[(size_t)b * 512 + h] = a;
    if (hout16) hout16[(size_t)b * 512 + h] = (_Float16)a;
    if (hid)   hid[(size_t)b * hidStride + h] = a;
    if (hid16) hid16[(size_t)b * hidStride + h] = (_Float16)a;
  }
}

// f32 -> f16 weight conversion; rows [rowsValid, gridDim.x) and k in
// [Kin, K32) are zero-filled (row/K padding for guard-free GEMM staging).
__global__ void cvtw_kernel(const float* __restrict__ src, _Float16* __restrict__ dst,
                            int rowsValid, int Kin, int K32) {
  size_t row = blockIdx.x;
  for (int k = threadIdx.x; k < K32; k += blockDim.x)
    dst[row * K32 + k] = (row < (size_t)rowsValid && k < Kin)
                       ? (_Float16)src[row * Kin + k] : (_Float16)0.f;
}

// Transposing variant: dst[n][k] = src[k][n]  (for q = h @ attn_w).
__global__ void cvtwT_kernel(const float* __restrict__ src, _Float16* __restrict__ dst,
                             int N, int K, int K32) {
  size_t nrow = blockIdx.x;
  for (int k = threadIdx.x; k < K32; k += blockDim.x)
    dst[nrow * K32 + k] = (k < K) ? (_Float16)src[(size_t)k * N + nrow] : (_Float16)0.f;
}

// flat f32 -> f16
__global__ void cvt_f2h_kernel(const float* __restrict__ s, _Float16* __restrict__ d, int n) {
  int i = blockIdx.x * blockDim.x + threadIdx.x;
  if (i < n) d[i] = (_Float16)s[i];
}

__global__ void zero_kernel(float* p, int n) {
  int i = blockIdx.x * blockDim.x + threadIdx.x;
  if (i < n) p[i] = 0.f;
}
__global__ void zeroh_kernel(_Float16* p, int n) {
  int i = blockIdx.x * blockDim.x + threadIdx.x;
  if (i < n) p[i] = (_Float16)0.f;
}

// ---------------------------------------------------------------------------
// Host orchestration
// ---------------------------------------------------------------------------
extern "C" void kernel_launch(void* const* d_in, const int* in_sizes, int n_in,
                              void* d_out, int out_size, void* d_ws, size_t ws_size,
                              hipStream_t stream) {
  (void)in_sizes; (void)n_in; (void)out_size; (void)ws_size;

  const float* x = (const float*)d_in[0];
  // d_in[1] = length (fixed 32 in harness)
  const float* vw[13]; const float* vb[13];
  for (int i = 0; i < 13; ++i) { vw[i] = (const float*)d_in[2 + i]; vb[i] = (const float*)d_in[15 + i]; }
  const float* det_w    = (const float*)d_in[28];
  const float* det_b    = (const float*)d_in[29];
  const float* cls_w    = (const float*)d_in[30];
  const float* cls_b    = (const float*)d_in[31];
  const float* enc_w_ih = (const float*)d_in[32];
  const float* enc_w_hh = (const float*)d_in[33];
  const float* enc_b_ih = (const float*)d_in[34];
  const float* enc_b_hh = (const float*)d_in[35];
  const float* dec_w_ih = (const float*)d_in[36];
  const float* dec_w_hh = (const float*)d_in[37];
  const float* dec_b_ih = (const float*)d_in[38];
  const float* dec_b_hh = (const float*)d_in[39];
  const float* embed_w  = (const float*)d_in[40];
  const float* embed_b  = (const float*)d_in[41];
  const float* attn_w   = (const float*)d_in[42];
  const float* box_w1   = (const float*)d_in[43];
  const float* box_b1   = (const float*)d_in[44];
  const float* box_w2   = (const float*)d_in[45];
  const float* box_b2   = (const float*)d_in[46];
  const float* cls_w1   = (const float*)d_in[47];
  const float* cls_b1   = (const float*)d_in[48];
  const float* cls_w2   = (const float*)d_in[49];
  const float* cls_b2   = (const float*)d_in[50];
  float* out = (float*)d_out;

  // ---- workspace carve-out ----
  char* base = (char*)d_ws;
  size_t off = 0;
  auto alloc = [&](size_t bytes) -> void* {
    off = (off + 255) & ~(size_t)255;
    void* p = base + off;
    off += bytes;
    return p;
  };
  _Float16* xh      = (_Float16*)alloc(4ull * 3 * 224 * 224 * 2);
  _Float16* actA    = (_Float16*)alloc(4ull * 64 * 224 * 224 * 2);
  _Float16* actB    = (_Float16*)alloc(4ull * 64 * 224 * 224 * 2);
  _Float16* detF    = (_Float16*)alloc(4ull * 128 * 112 * 112 * 2);
  _Float16* poolDet = (_Float16*)alloc(4ull * 128 * 56 * 56 * 2);
  _Float16* poolCls = (_Float16*)alloc(4ull * 512 * 7 * 7 * 2);
  _Float16* combined16 = (_Float16*)alloc(256ull * 3264 * 2);   // (B*64) x 3264
  float* xproj    = (float*)alloc(4ull * 64 * 1536 * 4);
  float* encOut   = (float*)alloc(4ull * 64 * 512 * 4);
  float* hEnc     = (float*)alloc(4 * 512 * 4);
  float* hDec     = (float*)alloc(4 * 512 * 4);
  float* hCell    = (float*)alloc(4 * 512 * 4);
  float* hpBuf    = (float*)alloc(4 * 1536 * 4);
  float* xpDec    = (float*)alloc(4 * 1536 * 4);
  float* qB       = (float*)alloc(4 * 512 * 4);
  float* hiddens  = (float*)alloc(4ull * 32 * 512 * 4);
  _Float16* hEnc16    = (_Float16*)alloc(32 * 512 * 2);   // 4 valid rows, rest zero
  _Float16* hDec16    = (_Float16*)alloc(32 * 512 * 2);
  _Float16* hCell16   = (_Float16*)alloc(32 * 512 * 2);
  _Float16* embB16    = (_Float16*)alloc(32 * 512 * 2);
  _Float16* hiddens16 = (_Float16*)alloc(128 * 512 * 2);
  _Float16* t116      = (_Float16*)alloc(128 * 512 * 2);

  static const int cinL[13]  = {3, 64, 64, 128, 128, 256, 256, 256, 512, 512, 512, 512, 512};
  static const int coutL[13] = {64, 64, 128, 128, 256, 256, 256, 512, 512, 512, 512, 512, 512};
  _Float16* w16[13];
  int K32L[13];
  for (int i = 0; i < 13; ++i) {
    int K = cinL[i] * 9;
    K32L[i] = (K + 31) & ~31;
    w16[i] = (_Float16*)alloc((size_t)coutL[i] * K32L[i] * 2);
  }
  _Float16* detW16   = (_Float16*)alloc(64 * 128 * 2);
  _Float16* clsW16   = (_Float16*)alloc(128 * 512 * 2);
  _Float16* encWih16 = (_Float16*)alloc(1536ull * 3264 * 2);
  _Float16* encWhh16 = (_Float16*)alloc(1536 * 512 * 2);
  _Float16* decWih16 = (_Float16*)alloc(1536 * 512 * 2);
  _Float16* decWhh16 = (_Float16*)alloc(1536 * 512 * 2);
  _Float16* embedW16 = (_Float16*)alloc(512 * 512 * 2);
  _Float16* attnWT16 = (_Float16*)alloc(512 * 512 * 2);
  _Float16* boxW116  = (_Float16*)alloc(512 * 512 * 2);
  _Float16* boxW216  = (_Float16*)alloc(64 * 512 * 2);    // 4 valid rows
  _Float16* clsW116  = (_Float16*)alloc(512 * 512 * 2);
  _Float16* clsW216  = (_Float16*)alloc(128 * 512 * 2);   // 100 valid rows

  // ---- weight + input conversion (once per launch; reused 96+ steps) ----
  for (int i = 0; i < 13; ++i)
    cvtw_kernel<<<coutL[i], 256, 0, stream>>>(vw[i], w16[i], coutL[i], cinL[i] * 9, K32L[i]);
  cvtw_kernel<<<64, 256, 0, stream>>>(det_w, detW16, 64, 128, 128);
  cvtw_kernel<<<128, 256, 0, stream>>>(cls_w, clsW16, 128, 512, 512);
  cvtw_kernel<<<1536, 256, 0, stream>>>(enc_w_ih, encWih16, 1536, 3234, 3264);
  cvtw_kernel<<<1536, 256, 0, stream>>>(enc_w_hh, encWhh16, 1536, 512, 512);
  cvtw_kernel<<<1536, 256, 0, stream>>>(dec_w_ih, decWih16, 1536, 512, 512);
  cvtw_kernel<<<1536, 256, 0, stream>>>(dec_w_hh, decWhh16, 1536, 512, 512);
  cvtw_kernel<<<512, 256, 0, stream>>>(embed_w, embedW16, 512, 512, 512);
  cvtwT_kernel<<<512, 256, 0, stream>>>(attn_w, attnWT16, 512, 512, 512);
  cvtw_kernel<<<512, 256, 0, stream>>>(box_w1, boxW116, 512, 512, 512);
  cvtw_kernel<<<64, 256, 0, stream>>>(box_w2, boxW216, 4, 512, 512);
  cvtw_kernel<<<512, 256, 0, stream>>>(cls_w1, clsW116, 512, 512, 512);
  cvtw_kernel<<<128, 256, 0, stream>>>(cls_w2, clsW216, 100, 512, 512);
  cvt_f2h_kernel<<<(602112 + 255) / 256, 256, 0, stream>>>(x, xh, 602112);

  // zero state + padded buffers (pad rows/cols must be zero)
  zero_kernel<<<8, 256, 0, stream>>>(hEnc, 2048);
  zero_kernel<<<8, 256, 0, stream>>>(hDec, 2048);
  zeroh_kernel<<<(256 * 3264 + 255) / 256, 256, 0, stream>>>(combined16, 256 * 3264);
  zeroh_kernel<<<64, 256, 0, stream>>>(hEnc16, 32 * 512);
  zeroh_kernel<<<64, 256, 0, stream>>>(hDec16, 32 * 512);
  zeroh_kernel<<<64, 256, 0, stream>>>(hCell16, 32 * 512);
  zeroh_kernel<<<64, 256, 0, stream>>>(embB16, 32 * 512);

  auto conv3 = [&](const _Float16* in, int li, _Float16* o, int H, int W) {
    dim3 g((H * W + 31) / 32, coutL[li] / 64, 4);
    conv3x3_wmma<<<g, 128, 0, stream>>>(in, w16[li], vb[li], o,
                                        cinL[li], coutL[li], H, W,
                                        cinL[li] * 9, K32L[li]);
  };
  auto pool = [&](const _Float16* in, _Float16* o, int C, int H, int W, int isMax) {
    int total = 4 * C * (H / 2) * (W / 2);
    pool2_kernel<<<(total + 255) / 256, 256, 0, stream>>>(in, o, C, H, W, isMax);
  };
  auto gemm = [&](const _Float16* A16, int lda16, const _Float16* W16, const float* bs,
                  float* Cf, int ldc, _Float16* C16, int ldc16,
                  int M, int N, int K32, int act) {
    dim3 g((M + 31) / 32, (N + 63) / 64);
    gemm16_wmma<<<g, 128, 0, stream>>>(A16, lda16, W16, bs, Cf, ldc, C16, ldc16, M, N, K32, act);
  };

  // ---- VGG16 backbone (f16 activations) ----
  conv3(xh,   0, actA, 224, 224);
  conv3(actA, 1, actB, 224, 224);
  pool(actB, actA, 64, 224, 224, 1);
  conv3(actA, 2, actB, 112, 112);
  conv3(actB, 3, detF, 112, 112);          // det_feat = conv2_2 output
  pool(detF, actA, 128, 112, 112, 1);
  conv3(actA, 4, actB, 56, 56);
  conv3(actB, 5, actA, 56, 56);
  conv3(actA, 6, actB, 56, 56);
  pool(actB, actA, 256, 56, 56, 1);
  conv3(actA, 7, actB, 28, 28);
  conv3(actB, 8, actA, 28, 28);
  conv3(actA, 9, actB, 28, 28);
  pool(actB, actA, 512, 28, 28, 1);
  conv3(actA, 10, actB, 14, 14);
  conv3(actB, 11, actA, 14, 14);
  conv3(actA, 12, actB, 14, 14);           // cls_feat in actB (4,512,14,14)

  // ---- projections into combined16 (B*64 x 3264, valid 3234) ----
  pool(detF, poolDet, 128, 112, 112, 0);   // avgpool -> (4,128,56,56)
  pool(actB, poolCls, 512, 14, 14, 1);     // maxpool -> (4,512,7,7)
  {
    dim3 g((3136 + 31) / 32, 1, 4);
    conv1x1_wmma<<<g, 128, 0, stream>>>(poolDet, detW16, det_b, combined16,
                                        128, 64, 3136, 128, 0);
  }
  {
    dim3 g((49 + 31) / 32, 2, 4);
    conv1x1_wmma<<<g, 128, 0, stream>>>(poolCls, clsW16, cls_b, combined16,
                                        512, 128, 49, 512, 1);
  }

  // ---- encoder: x_proj then 64 GRU steps ----
  gemm(combined16, 3264, encWih16, enc_b_ih, xproj, 1536, nullptr, 0, 256, 1536, 3264, 0);
  for (int s = 0; s < 64; ++s) {
    gemm(hEnc16, 512, encWhh16, enc_b_hh, hpBuf, 1536, nullptr, 0, 4, 1536, 512, 0);
    gru_kernel<<<8, 256, 0, stream>>>(xproj + s * 1536, 64L * 1536, hpBuf,
                                      hEnc, hEnc, hEnc16, encOut + s * 512, 64L * 512);
  }

  // ---- decoder: 32 steps with Luong attention ----
  gemm(hEnc16, 512, decWih16, dec_b_ih, xpDec, 1536, nullptr, 0, 4, 1536, 512, 0);
  gemm(hDec16, 512, decWhh16, dec_b_hh, hpBuf, 1536, nullptr, 0, 4, 1536, 512, 0);  // hDec16 == 0
  gru_kernel<<<8, 256, 0, stream>>>(xpDec, 1536, hpBuf, hDec, hCell, hCell16, nullptr, 0);
  gemm(hCell16, 512, attnWT16, nullptr, qB, 512, nullptr, 0, 4, 512, 512, 0);
  attn_kernel<<<4, 256, 0, stream>>>(qB, encOut, hDec, hDec16,
                                     hiddens + 0, hiddens16 + 0, 32L * 512);

  for (int t = 1; t < 32; ++t) {
    gemm(hDec16, 512, embedW16, embed_b, nullptr, 0, embB16, 512, 4, 512, 512, 0);
    gemm(embB16, 512, decWih16, dec_b_ih, xpDec, 1536, nullptr, 0, 4, 1536, 512, 0);
    gemm(hDec16, 512, decWhh16, dec_b_hh, hpBuf, 1536, nullptr, 0, 4, 1536, 512, 0);
    gru_kernel<<<8, 256, 0, stream>>>(xpDec, 1536, hpBuf, hDec, hCell, hCell16, nullptr, 0);
    gemm(hCell16, 512, attnWT16, nullptr, qB, 512, nullptr, 0, 4, 512, 512, 0);
    attn_kernel<<<4, 256, 0, stream>>>(qB, encOut, hDec, hDec16,
                                       hiddens + t * 512, hiddens16 + t * 512, 32L * 512);
  }

  // ---- output heads ----
  gemm(hiddens16, 512, boxW116, box_b1, nullptr, 0, t116, 512, 128, 512, 512, 1);  // relu
  gemm(t116, 512, boxW216, box_b2, out, 4, nullptr, 0, 128, 4, 512, 2);            // sigmoid
  gemm(hiddens16, 512, clsW116, cls_b1, nullptr, 0, t116, 512, 128, 512, 512, 1);  // relu
  gemm(t116, 512, clsW216, cls_b2, out + 512, 100, nullptr, 0, 128, 100, 512, 0);  // logits
}
